// Attn_22067541966907
// MI455X (gfx1250) — matrix operand order repaired and checked by reading, not compiled, not run on hardware
//
#include <hip/hip_runtime.h>

// ---------------------------------------------------------------------------
// HEPT attention block, MI455X (gfx1250, wave32, WMMA).
// bf16 WMMA everywhere heavy; fp32 softmax/LN; async global->LDS staging
// (double-buffered) for K/V tiles in the flash-attention kernel.
// ---------------------------------------------------------------------------

typedef __bf16 bf16;
typedef __attribute__((ext_vector_type(4)))  __bf16 v4bf;
typedef __attribute__((ext_vector_type(8)))  __bf16 v8bf;
typedef __attribute__((ext_vector_type(16))) __bf16 v16bf;
typedef __attribute__((ext_vector_type(8)))  float  v8f;
typedef __attribute__((ext_vector_type(4)))  float  v4f;

#define NPTS 4096
#define DH   128
#define NH   8
#define HD   1024   /* NH*DH */

__device__ __forceinline__ v16bf make_v16(v8bf lo, v8bf hi) {
  return __builtin_shufflevector(lo, hi, 0,1,2,3,4,5,6,7,8,9,10,11,12,13,14,15);
}

__device__ __forceinline__ v8f wmma_bf16(v16bf a, v16bf b, v8f c) {
  return __builtin_amdgcn_wmma_f32_16x16x32_bf16(false, a, false, b, (short)0, c,
                                                 false, false);
}

// A fragment, 16x32 tile of row-major A (lda elements).
// This code's k order (B uses the same): lane L holds row m=L&15, K = 16*(L>=16)+(0..15) contiguous.
__device__ __forceinline__ v16bf load_a_frag(const bf16* __restrict__ A, int lda,
                                             int row0, int k0, int lane) {
  int m = lane & 15, hi = lane >> 4;
  const bf16* p = A + (size_t)(row0 + m) * lda + k0 + 16 * hi;
  v8bf lo = *(const v8bf*)p;
  v8bf hs = *(const v8bf*)(p + 8);
  return make_v16(lo, hs);
}

// B fragment, 32x16 tile from transposed storage Bt[n,k] (row-major):
// lane L holds column n=L&15, K = 16*(L>=16)+(0..15) contiguous -> 32B load.
__device__ __forceinline__ v16bf load_b_fragT(const bf16* __restrict__ Bt, int ldt,
                                              int n0, int k0, int lane) {
  int n = lane & 15, hi = lane >> 4;
  const bf16* p = Bt + (size_t)(n0 + n) * ldt + k0 + 16 * hi;
  return *(const v16bf*)p;
}

// Async global->LDS 16-byte copy (per lane), CDNA5 TDM-adjacent path.
// Tracked by ASYNCcnt; completed with s_wait_asynccnt.
__device__ __forceinline__ void async_copy16(unsigned lds_off, const void* g) {
  unsigned long long ga = (unsigned long long)(uintptr_t)g;
  asm volatile("global_load_async_to_lds_b128 %0, %1, off"
               :: "v"(lds_off), "v"(ga) : "memory");
}
__device__ __forceinline__ void wait_async0() {
  asm volatile("s_wait_asynccnt 0x0" ::: "memory");
}

// ---------------------------------------------------------------------------
// rpe weight reduction:  w[h,c] = mean_{d,wi} relu(rpe_w[h*128+d, c*8+wi])
// ---------------------------------------------------------------------------
__global__ void rpe_reduce(const float* __restrict__ rpe, float* __restrict__ w16) {
  int lane = threadIdx.x & 31;
  int o    = threadIdx.x >> 5;      // blockDim = 512 -> 16 outputs
  int h = o >> 1, c = o & 1;
  float acc = 0.0f;
  for (int d = lane; d < DH; d += 32) {
    const float* row = rpe + (size_t)(h * DH + d) * 16 + c * 8;
#pragma unroll
    for (int wi = 0; wi < 8; ++wi) acc += fmaxf(row[wi], 0.0f);
  }
#pragma unroll
  for (int m = 16; m >= 1; m >>= 1) acc += __shfl_xor(acc, m, 32);
  if (lane == 0) w16[o] = acc * (1.0f / 1024.0f);
}

// sdist[i*8+h] = w[h,0]*c1^2 + w[h,1]*c2^2   (rank-2 bias per-point term)
__global__ void coord_prep(const float* __restrict__ coords,
                           const float* __restrict__ w16,
                           float* __restrict__ sdist) {
  int idx = blockIdx.x * blockDim.x + threadIdx.x;
  if (idx >= NPTS * NH) return;
  int i = idx >> 3, h = idx & 7;
  float c1 = coords[i * 3 + 1], c2 = coords[i * 3 + 2];
  sdist[idx] = w16[h * 2 + 0] * c1 * c1 + w16[h * 2 + 1] * c2 * c2;
}

// ---------------------------------------------------------------------------
// LayerNorm (one wave per 128-wide row) -> bf16
// ---------------------------------------------------------------------------
__global__ void layernorm_to_bf16(const float* __restrict__ x,
                                  const float* __restrict__ g,
                                  const float* __restrict__ b,
                                  bf16* __restrict__ out, int nrows) {
  int lane = threadIdx.x & 31;
  int wave = threadIdx.x >> 5;
  int row  = blockIdx.x * (blockDim.x >> 5) + wave;
  if (row >= nrows) return;
  const float* xr = x + (size_t)row * DH;
  v4f v = *(const v4f*)(xr + lane * 4);
  float s = v[0] + v[1] + v[2] + v[3];
#pragma unroll
  for (int m = 16; m >= 1; m >>= 1) s += __shfl_xor(s, m, 32);
  float mu = s * (1.0f / DH);
  float d0 = v[0] - mu, d1 = v[1] - mu, d2 = v[2] - mu, d3 = v[3] - mu;
  float q = d0 * d0 + d1 * d1 + d2 * d2 + d3 * d3;
#pragma unroll
  for (int m = 16; m >= 1; m >>= 1) q += __shfl_xor(q, m, 32);
  float inv = rsqrtf(q * (1.0f / DH) + 1e-5f);
  float dd[4] = {d0, d1, d2, d3};
  v4bf o;
#pragma unroll
  for (int j = 0; j < 4; ++j) {
    int c = lane * 4 + j;
    o[j] = (bf16)(dd[j] * inv * g[c] + b[c]);
  }
  *(v4bf*)(out + (size_t)row * DH + lane * 4) = o;
}

// fp32 [K,N] row-major  ->  bf16 [N,K] row-major (transposed weights)
__global__ void convert_transpose(const float* __restrict__ src,
                                  bf16* __restrict__ dst, int K, int N) {
  int idx = blockIdx.x * blockDim.x + threadIdx.x;
  if (idx >= K * N) return;
  int k = idx / N, n = idx % N;
  dst[(size_t)n * K + k] = (bf16)src[idx];
}

// ---------------------------------------------------------------------------
// WMMA GEMM: C[M,N] = A[M,K](bf16) * Bt[N,K](bf16)^T
// One 16x64 tile per wave (A-fragment reused across 4 B-fragments).
// Epilogue: +bias, relu, +resid(f32), store f32 / bf16 / bf16-transposed.
// ---------------------------------------------------------------------------
__global__ void __launch_bounds__(256) gemm_wmma_bf16(
    const bf16* __restrict__ A, int lda,
    const bf16* __restrict__ Bt, int ldt,
    int M, int N, int K,
    const float* __restrict__ bias,
    const float* __restrict__ resid,
    float* __restrict__ outF,
    bf16* __restrict__ outB,
    bf16* __restrict__ outBT,
    int relu) {
  int lane = threadIdx.x & 31;
  int wave = threadIdx.x >> 5;
  int tile = blockIdx.x * 8 + wave;
  int ntn  = N >> 6;                        // 64-wide n blocks
  int tm = tile / ntn, tn = tile % ntn;
  if (tm * 16 >= M) return;                 // wave-uniform guard
  int row0 = tm * 16, n0 = tn * 64;

  v8f acc[4] = {};
  for (int k0 = 0; k0 < K; k0 += 32) {
    v16bf a = load_a_frag(A, lda, row0, k0, lane);
#pragma unroll
    for (int nb = 0; nb < 4; ++nb) {
      v16bf b = load_b_fragT(Bt, ldt, n0 + nb * 16, k0, lane);
      acc[nb] = wmma_bf16(a, b, acc[nb]);
    }
  }

  int hi = lane >> 4;
#pragma unroll
  for (int nb = 0; nb < 4; ++nb) {
    int col = n0 + nb * 16 + (lane & 15);
    float bv = bias ? bias[col] : 0.0f;
    v8bf packed;
#pragma unroll
    for (int r = 0; r < 8; ++r) {
      int row = row0 + r + 8 * hi;
      float v = acc[nb][r] + bv;
      if (relu) v = fmaxf(v, 0.0f);
      if (resid) v += resid[(size_t)row * N + col];
      if (outF) outF[(size_t)row * N + col] = v;
      if (outB) outB[(size_t)row * N + col] = (bf16)v;
      packed[r] = (bf16)v;
    }
    if (outBT)   // column 'col' of C, rows row0+8*hi..+7 contiguous
      *(v8bf*)(outBT + (size_t)col * M + row0 + 8 * hi) = packed;
  }
}

// ---------------------------------------------------------------------------
// Flash attention.  Block = 256 threads (8 waves) = one head, 128 query rows.
// 64-key chunks of K/V staged into LDS with async b128 loads, double-buffered.
// logits = (Q.K^T)/sqrt(D) - (s_i + s_j - 2*(w*c_i).c_j)
// ---------------------------------------------------------------------------
__device__ __forceinline__ void stage_kv(int tid, int head, int jb,
                                         const bf16* __restrict__ kbuf,
                                         const bf16* __restrict__ vT,
                                         bf16 (*kS)[DH], bf16 (*vS)[64]) {
  if (tid < 128) {                  // K tile: 64 rows x 128 d  (16 KB)
    int j = tid >> 1, half = tid & 1;
    const bf16* src = kbuf + (size_t)(jb + j) * HD + head * DH + half * 64;
    unsigned dst = (unsigned)(uintptr_t)&kS[j][half * 64];
#pragma unroll
    for (int i = 0; i < 8; ++i) async_copy16(dst + i * 16, src + i * 8);
  } else {                          // V tile: 128 d-rows x 64 keys (16 KB)
    int d = tid - 128;
    const bf16* src = vT + (size_t)(head * DH + d) * NPTS + jb;
    unsigned dst = (unsigned)(uintptr_t)&vS[d][0];
#pragma unroll
    for (int i = 0; i < 8; ++i) async_copy16(dst + i * 16, src + i * 8);
  }
}

__global__ void __launch_bounds__(256) attn_flash(
    const bf16* __restrict__ qbuf,   // [NPTS, HD]
    const bf16* __restrict__ kbuf,   // [NPTS, HD]
    const bf16* __restrict__ vT,     // [HD, NPTS]
    const float* __restrict__ coords,
    const float* __restrict__ sdist, // [NPTS, NH]
    const float* __restrict__ w16,   // [NH, 2]
    bf16* __restrict__ attn_out) {   // [NPTS, HD]
  __shared__ __align__(128) bf16 kS[2][64][DH];   // 2 x 16 KB
  __shared__ __align__(128) bf16 vS[2][DH][64];   // 2 x 16 KB
  __shared__ __align__(64)  bf16 pS[8][16][64];   // 16 KB

  int tid  = threadIdx.x;
  int lane = tid & 31, wave = tid >> 5;
  int m = lane & 15, hi = lane >> 4;
  int head = blockIdx.x >> 5;        // 32 blocks per head
  int qblk = blockIdx.x & 31;
  int row0 = qblk * 128 + wave * 16;
  const float scale = 0.08838834764831845f; // 1/sqrt(128)
  float w1 = w16[head * 2 + 0], w2 = w16[head * 2 + 1];

  v16bf qf[4];
#pragma unroll
  for (int kd = 0; kd < 4; ++kd)
    qf[kd] = load_a_frag(qbuf, HD, row0, head * DH + kd * 32, lane);

  float sq[8], pq1[8], pq2[8], mrow[8], lrow[8];
#pragma unroll
  for (int r = 0; r < 8; ++r) {
    int i  = row0 + r + 8 * hi;
    sq[r]  = sdist[i * NH + head];
    pq1[r] = w1 * coords[i * 3 + 1];
    pq2[r] = w2 * coords[i * 3 + 2];
    mrow[r] = -1e30f; lrow[r] = 0.0f;
  }
  v8f o[8];
#pragma unroll
  for (int dc = 0; dc < 8; ++dc) o[dc] = (v8f){};

  stage_kv(tid, head, 0, kbuf, vT, kS[0], vS[0]);
  wait_async0();
  __syncthreads();

  for (int c = 0; c < NPTS / 64; ++c) {
    int jb = c * 64;
    int cur = c & 1;
    if (c + 1 < NPTS / 64)           // prefetch next chunk into other buffer
      stage_kv(tid, head, jb + 64, kbuf, vT, kS[cur ^ 1], vS[cur ^ 1]);

    // S = Q.K^T over 4 key subtiles (16 WMMA)
    v8f s[4] = {};
#pragma unroll
    for (int ts = 0; ts < 4; ++ts)
#pragma unroll
      for (int kd = 0; kd < 4; ++kd) {
        v8bf b0 = *(const v8bf*)&kS[cur][ts * 16 + m][kd * 32 + 16 * hi];
        v8bf b1 = *(const v8bf*)&kS[cur][ts * 16 + m][kd * 32 + 16 * hi + 8];
        s[ts] = wmma_bf16(qf[kd], make_v16(b0, b1), s[ts]);
      }

    float c1[4], c2[4], sj[4];
#pragma unroll
    for (int ts = 0; ts < 4; ++ts) {
      int j = jb + ts * 16 + m;
      c1[ts] = coords[j * 3 + 1];
      c2[ts] = coords[j * 3 + 2];
      sj[ts] = sdist[j * NH + head];
    }

    // online softmax over 64 columns
#pragma unroll
    for (int r = 0; r < 8; ++r) {
      float l[4];
#pragma unroll
      for (int ts = 0; ts < 4; ++ts)
        l[ts] = s[ts][r] * scale -
                (sq[r] + sj[ts] - 2.0f * (pq1[r] * c1[ts] + pq2[r] * c2[ts]));
      float mx = fmaxf(fmaxf(l[0], l[1]), fmaxf(l[2], l[3]));
#pragma unroll
      for (int sh = 8; sh >= 1; sh >>= 1) mx = fmaxf(mx, __shfl_xor(mx, sh, 32));
      float mnew = fmaxf(mrow[r], mx);
      float al = __expf(mrow[r] - mnew);
      float ps = 0.0f;
#pragma unroll
      for (int ts = 0; ts < 4; ++ts) {
        float p = __expf(l[ts] - mnew);
        s[ts][r] = p;
        ps += p;
      }
#pragma unroll
      for (int sh = 8; sh >= 1; sh >>= 1) ps += __shfl_xor(ps, sh, 32);
      lrow[r] = lrow[r] * al + ps;
      mrow[r] = mnew;
#pragma unroll
      for (int dc = 0; dc < 8; ++dc) o[dc][r] *= al;
    }

    // stage P (C-layout) -> LDS -> reload in A-layout (same-wave, in-order LDS)
#pragma unroll
    for (int ts = 0; ts < 4; ++ts)
#pragma unroll
      for (int r = 0; r < 8; ++r)
        pS[wave][r + 8 * hi][ts * 16 + m] = (bf16)s[ts][r];
    asm volatile("" ::: "memory");

    // O += P.V  (16 WMMA)
#pragma unroll
    for (int js = 0; js < 2; ++js) {
      v8bf plo = *(const v8bf*)&pS[wave][m][js * 32 + 16 * hi];
      v8bf phi = *(const v8bf*)&pS[wave][m][js * 32 + 16 * hi + 8];
      v16bf pa = make_v16(plo, phi);
#pragma unroll
      for (int dc = 0; dc < 8; ++dc) {
        v8bf v0 = *(const v8bf*)&vS[cur][dc * 16 + m][js * 32 + 16 * hi];
        v8bf v1 = *(const v8bf*)&vS[cur][dc * 16 + m][js * 32 + 16 * hi + 8];
        o[dc] = wmma_bf16(pa, make_v16(v0, v1), o[dc]);
      }
    }
    asm volatile("" ::: "memory");
    wait_async0();       // own async stores to the other buffer done
    __syncthreads();     // everyone done reading cur + staging other
  }

#pragma unroll
  for (int r = 0; r < 8; ++r) {
    float invl = 1.0f / lrow[r];
    int row = row0 + r + 8 * hi;
#pragma unroll
    for (int dc = 0; dc < 8; ++dc)
      attn_out[(size_t)row * HD + head * DH + dc * 16 + m] = (bf16)(o[dc][r] * invl);
  }
}

// ---------------------------------------------------------------------------
extern "C" void kernel_launch(void* const* d_in, const int* in_sizes, int n_in,
                              void* d_out, int out_size, void* d_ws, size_t ws_size,
                              hipStream_t stream) {
  const float* x      = (const float*)d_in[0];
  const float* coords = (const float*)d_in[1];
  const float* ln1_g  = (const float*)d_in[2];
  const float* ln1_b  = (const float*)d_in[3];
  const float* wq     = (const float*)d_in[4];
  const float* wk     = (const float*)d_in[5];
  const float* wv     = (const float*)d_in[6];
  const float* rpe_w  = (const float*)d_in[7];
  const float* wo_w   = (const float*)d_in[8];
  const float* wo_b   = (const float*)d_in[9];
  const float* ln2_g  = (const float*)d_in[10];
  const float* ln2_b  = (const float*)d_in[11];
  const float* ff_w1  = (const float*)d_in[12];
  const float* ff_b1  = (const float*)d_in[13];
  const float* ff_w2  = (const float*)d_in[14];
  const float* ff_b2  = (const float*)d_in[15];
  float* out = (float*)d_out;

  char* ws = (char*)d_ws;
  size_t off = 0;
  auto alloc = [&](size_t bytes) -> void* {
    void* p = ws + off;
    off = (off + bytes + 255) & ~(size_t)255;
    return p;
  };

  bf16*  xn   = (bf16*)alloc((size_t)NPTS * DH * 2);
  bf16*  wqT  = (bf16*)alloc((size_t)DH * HD * 2);
  bf16*  wkT  = (bf16*)alloc((size_t)DH * HD * 2);
  bf16*  wvT  = (bf16*)alloc((size_t)DH * HD * 2);
  bf16*  woT  = (bf16*)alloc((size_t)HD * DH * 2);
  bf16*  f1T  = (bf16*)alloc((size_t)DH * DH * 2);
  bf16*  f2T  = (bf16*)alloc((size_t)DH * DH * 2);
  bf16*  qb   = (bf16*)alloc((size_t)NPTS * HD * 2);
  bf16*  kb   = (bf16*)alloc((size_t)NPTS * HD * 2);
  bf16*  vT   = (bf16*)alloc((size_t)HD * NPTS * 2);
  bf16*  ao   = (bf16*)alloc((size_t)NPTS * HD * 2);
  float* x2   = (float*)alloc((size_t)NPTS * DH * 4);
  bf16*  h2   = (bf16*)alloc((size_t)NPTS * DH * 2);
  bf16*  ffh  = (bf16*)alloc((size_t)NPTS * DH * 2);
  float* w16  = (float*)alloc(64);
  float* sd   = (float*)alloc((size_t)NPTS * NH * 4);

  rpe_reduce<<<1, 512, 0, stream>>>(rpe_w, w16);
  coord_prep<<<(NPTS * NH + 255) / 256, 256, 0, stream>>>(coords, w16, sd);

  layernorm_to_bf16<<<NPTS / 8, 256, 0, stream>>>(x, ln1_g, ln1_b, xn, NPTS);

  convert_transpose<<<(DH * HD + 255) / 256, 256, 0, stream>>>(wq, wqT, DH, HD);
  convert_transpose<<<(DH * HD + 255) / 256, 256, 0, stream>>>(wk, wkT, DH, HD);
  convert_transpose<<<(DH * HD + 255) / 256, 256, 0, stream>>>(wv, wvT, DH, HD);
  convert_transpose<<<(HD * DH + 255) / 256, 256, 0, stream>>>(wo_w, woT, HD, DH);
  convert_transpose<<<(DH * DH + 255) / 256, 256, 0, stream>>>(ff_w1, f1T, DH, DH);
  convert_transpose<<<(DH * DH + 255) / 256, 256, 0, stream>>>(ff_w2, f2T, DH, DH);

  auto gemm_blocks = [](int M, int N) { return ((M / 16) * (N / 64) + 7) / 8; };

  gemm_wmma_bf16<<<gemm_blocks(NPTS, HD), 256, 0, stream>>>(
      xn, DH, wqT, DH, NPTS, HD, DH, nullptr, nullptr, nullptr, qb, nullptr, 0);
  gemm_wmma_bf16<<<gemm_blocks(NPTS, HD), 256, 0, stream>>>(
      xn, DH, wkT, DH, NPTS, HD, DH, nullptr, nullptr, nullptr, kb, nullptr, 0);
  gemm_wmma_bf16<<<gemm_blocks(NPTS, HD), 256, 0, stream>>>(
      xn, DH, wvT, DH, NPTS, HD, DH, nullptr, nullptr, nullptr, nullptr, vT, 0);

  // 8 heads * 32 q-blocks (128 rows each) = 256 blocks of 8 waves
  attn_flash<<<256, 256, 0, stream>>>(qb, kb, vT, coords, sd, w16, ao);

  gemm_wmma_bf16<<<gemm_blocks(NPTS, DH), 256, 0, stream>>>(
      ao, HD, woT, HD, NPTS, DH, HD, wo_b, x, x2, nullptr, nullptr, 0);

  layernorm_to_bf16<<<NPTS / 8, 256, 0, stream>>>(x2, ln2_g, ln2_b, h2, NPTS);

  gemm_wmma_bf16<<<gemm_blocks(NPTS, DH), 256, 0, stream>>>(
      h2, DH, f1T, DH, NPTS, DH, DH, ff_b1, nullptr, nullptr, ffh, nullptr, 1);

  gemm_wmma_bf16<<<gemm_blocks(NPTS, DH), 256, 0, stream>>>(
      ffh, DH, f2T, DH, NPTS, DH, DH, ff_b2, x2, out, nullptr, nullptr, 0);
}
